// DMPNNEncoder_86414741995742
// MI455X (gfx1250) — matrix operand order repaired and checked by reading, not compiled, not run on hardware
//
#include <hip/hip_runtime.h>

// ---------------------------------------------------------------------------
// D-MPNN encoder for MI455X (gfx1250, wave32, WMMA).
// GEMMs: C = relu(A_bf16 [MxK] @ W_bf16 [NxK]^T (+bias)) via
// v_wmma_f32_16x16x32_bf16, with double-buffered async global->LDS staging
// (global_load_async_to_lds_b128 + s_wait_asynccnt).
// ---------------------------------------------------------------------------

#define N_ATOMS 100000
#define N_BONDS 200000
#define N_MOLS  4000
#define MAX_NB  6
#define ATOM_FDIM 133
#define BOND_FDIM 14
#define HIDDEN 300

// padded dims (multiples of 32 for WMMA K, 64 for N tiling)
#define K1P 160   // 133+14=147 -> 160
#define HP  320   // 300 -> 320 (both as N-pad and K-pad)
#define K3P 448   // 133+300=433 -> 448

typedef __attribute__((ext_vector_type(16))) __bf16 v16bf;
typedef __attribute__((ext_vector_type(8)))  float  v8f;

union BV8 { uint4 u; __bf16 h[8]; };

// ---- CDNA5 async global->LDS copy (ASYNCcnt-tracked) ----------------------
__device__ __forceinline__ void async_ld_b128(unsigned lds_addr,
                                              unsigned long long gaddr) {
  asm volatile("global_load_async_to_lds_b128 %0, %1, off"
               :: "v"(lds_addr), "v"(gaddr)
               : "memory");
}
__device__ __forceinline__ void wait_async_le3() {
  asm volatile("s_wait_asynccnt 0x3" ::: "memory");
}
__device__ __forceinline__ void wait_async_0() {
  asm volatile("s_wait_asynccnt 0x0" ::: "memory");
}

// ---------------------------------------------------------------------------
// Weight pad+convert: out[orows][ocols] = bf16(W[r][c]) zero-padded
// ---------------------------------------------------------------------------
__global__ void pad_weights_bf16(const float* __restrict__ W, int rows, int cols,
                                 __bf16* __restrict__ out, int orows, int ocols) {
  int idx = blockIdx.x * blockDim.x + threadIdx.x;
  int total = orows * ocols;
  if (idx >= total) return;
  int r = idx / ocols;
  int c = idx - r * ocols;
  float v = (r < rows && c < cols) ? W[(size_t)r * cols + c] : 0.0f;
  out[idx] = (__bf16)v;
}

// ---------------------------------------------------------------------------
// X0[b][0:133]=f_atoms[b2a[b]], [133:147]=f_bonds[b], rest 0   (bf16)
// ---------------------------------------------------------------------------
__global__ void build_x0(const float* __restrict__ f_atoms,
                         const float* __restrict__ f_bonds,
                         const int* __restrict__ b2a,
                         __bf16* __restrict__ X0) {
  int idx = blockIdx.x * blockDim.x + threadIdx.x;
  if (idx >= N_BONDS * K1P) return;
  int b = idx / K1P;
  int c = idx - b * K1P;
  float v = 0.0f;
  if (c < ATOM_FDIM) {
    v = f_atoms[(size_t)b2a[b] * ATOM_FDIM + c];
  } else if (c < ATOM_FDIM + BOND_FDIM) {
    v = f_bonds[(size_t)b * BOND_FDIM + (c - ATOM_FDIM)];
  }
  X0[idx] = (__bf16)v;
}

// ---------------------------------------------------------------------------
// X1[b][k] = sum_j msg[a2b[b2a[b]][j]][k] - msg[b2revb[b]][k]      (bf16)
// 8 bf16 per thread: one b128 load per gathered row.
// ---------------------------------------------------------------------------
__global__ void build_x1(const __bf16* __restrict__ msg,
                         const int* __restrict__ a2b,
                         const int* __restrict__ b2a,
                         const int* __restrict__ b2revb,
                         __bf16* __restrict__ X1) {
  int idx = blockIdx.x * blockDim.x + threadIdx.x;
  if (idx >= N_BONDS * (HP / 8)) return;
  int b = idx / (HP / 8);
  int k8 = (idx - b * (HP / 8)) * 8;
  int a = b2a[b];
  float s[8] = {};
#pragma unroll
  for (int j = 0; j < MAX_NB; ++j) {
    BV8 v;
    v.u = *(const uint4*)(msg + (size_t)a2b[a * MAX_NB + j] * HP + k8);
#pragma unroll
    for (int e = 0; e < 8; ++e) s[e] += (float)v.h[e];
  }
  BV8 r;
  r.u = *(const uint4*)(msg + (size_t)b2revb[b] * HP + k8);
  BV8 o;
#pragma unroll
  for (int e = 0; e < 8; ++e) o.h[e] = (__bf16)(s[e] - (float)r.h[e]);
  *(uint4*)(X1 + (size_t)b * HP + k8) = o.u;
}

// ---------------------------------------------------------------------------
// X2[a][0:133]=f_atoms[a], [133:433]=sum_j msg[a2b[a][j]][0:300], rest 0
// ---------------------------------------------------------------------------
__global__ void build_x2(const float* __restrict__ f_atoms,
                         const __bf16* __restrict__ msg,
                         const int* __restrict__ a2b,
                         __bf16* __restrict__ X2) {
  int idx = blockIdx.x * blockDim.x + threadIdx.x;
  if (idx >= N_ATOMS * K3P) return;
  int a = idx / K3P;
  int c = idx - a * K3P;
  float v = 0.0f;
  if (c < ATOM_FDIM) {
    v = f_atoms[(size_t)a * ATOM_FDIM + c];
  } else if (c < ATOM_FDIM + HIDDEN) {
    int k = c - ATOM_FDIM;
#pragma unroll
    for (int j = 0; j < MAX_NB; ++j)
      v += (float)msg[(size_t)a2b[a * MAX_NB + j] * HP + k];
  }
  X2[idx] = (__bf16)v;
}

// ---------------------------------------------------------------------------
// Per-molecule mean of atom_hiddens (fp32 [N_ATOMS][300]) -> out [N_MOLS][300]
// ---------------------------------------------------------------------------
__global__ void mol_mean(const float* __restrict__ hidden,
                         const int* __restrict__ a_scope,
                         float* __restrict__ out) {
  int m = blockIdx.x;
  int start = a_scope[m * 2 + 0];
  int size  = a_scope[m * 2 + 1];
  for (int c = threadIdx.x; c < HIDDEN; c += blockDim.x) {
    float s = 0.0f;
    for (int i = 0; i < size; ++i)
      s += hidden[(size_t)(start + i) * HIDDEN + c];
    out[(size_t)m * HIDDEN + c] = (size > 0) ? s / (float)size : 0.0f;
  }
}

// ---------------------------------------------------------------------------
// WMMA GEMM: C = relu(A[MxK] @ W[NpadxK]^T (+bias))
//   256 threads = 8 wave32s; block tile 128x64; wave tile 32x32 (2x2 WMMA).
//   Double-buffered async global->LDS staging; LDS row stride padded to 40
//   bf16 for bank-conflict-free b128 fragment reads (ISA 7.12.2 layouts).
//   Per wave: 3 async instructions per staging group (2xA + 1xB), so
//   s_wait_asynccnt 3 drains the previous group after issuing the next.
//   A rows >= M are clamped to row M-1: garbage only reaches C rows >= M,
//   which the epilogue never stores (keeps ASYNCcnt uniform per wave).
// ---------------------------------------------------------------------------
template <bool F32OUT>
__global__ __launch_bounds__(256) void gemm_wmma(
    const __bf16* __restrict__ A, int K,
    const __bf16* __restrict__ W,
    int M,
    __bf16* __restrict__ Cb, int Npad,
    float* __restrict__ Cf, const float* __restrict__ bias, int Nreal) {
  __shared__ __align__(16) __bf16 As[2][128][40];
  __shared__ __align__(16) __bf16 Bs[2][64][40];

  const int tid  = threadIdx.x;
  const int m0   = blockIdx.x * 128;
  const int n0   = blockIdx.y * 64;
  const int lane = tid & 31;
  const int wave = tid >> 5;
  const int wm   = (wave & 3) * 32;   // wave M offset in block tile
  const int wn   = (wave >> 2) * 32;  // wave N offset in block tile
  const int l16  = lane & 15;
  const int lh   = lane >> 4;

  // per-thread staging assignment (fixed across K loop)
  const int arow = tid >> 1;
  const int aseg = (tid & 1) * 16;                 // bf16 elements
  const int gma  = (m0 + arow < M) ? (m0 + arow) : (M - 1);
  const unsigned long long gA =
      (unsigned long long)(uintptr_t)(A + (size_t)gma * K + aseg);
  const int brow = tid >> 2;
  const int bseg = (tid & 3) * 8;
  const unsigned long long gW =
      (unsigned long long)(uintptr_t)(W + (size_t)(n0 + brow) * K + bseg);

  auto stage = [&](int buf, int k0) {
    unsigned la = (unsigned)(uintptr_t)(&As[buf][arow][aseg]);
    unsigned long long ga = gA + (unsigned long long)k0 * 2;
    async_ld_b128(la, ga);
    async_ld_b128(la + 16u, ga + 16ull);
    unsigned lb = (unsigned)(uintptr_t)(&Bs[buf][brow][bseg]);
    async_ld_b128(lb, gW + (unsigned long long)k0 * 2);
  };

  v8f acc[2][2] = {};

  stage(0, 0);
  const int nk = K / 32;
  for (int kb = 0; kb < nk; ++kb) {
    const int cur = kb & 1;
    __syncthreads();  // (a) everyone done reading the buffer we stage next
    const bool has_next = (kb + 1 < nk);
    if (has_next) stage(cur ^ 1, (kb + 1) * 32);
    if (has_next) wait_async_le3(); else wait_async_0();
    __syncthreads();  // (b) buffer `cur` fully written by all waves

    union Frag { v16bf v; uint4 u[2]; };
    Frag fa[2], fb[2];
    // A frag: lane holds row (wm+i*16+l16); lanes0-15: K{0..7,16..23},
    // lanes16-31: K{8..15,24..31}  -> two b128 reads
#pragma unroll
    for (int i = 0; i < 2; ++i) {
      const __bf16* base = &As[cur][wm + i * 16 + l16][lh * 8];
      fa[i].u[0] = *(const uint4*)(base);
      fa[i].u[1] = *(const uint4*)(base + 16);
    }
    // B frag: lane holds col n=(wn+j*16+l16); lanes0-15: K{0..7,16..23},
    // lanes16-31: K{8..15,24..31}
#pragma unroll
    for (int j = 0; j < 2; ++j) {
      const __bf16* base = &Bs[cur][wn + j * 16 + l16][lh * 8];
      fb[j].u[0] = *(const uint4*)(base);
      fb[j].u[1] = *(const uint4*)(base + 16);
    }
#pragma unroll
    for (int i = 0; i < 2; ++i)
#pragma unroll
      for (int j = 0; j < 2; ++j)
        acc[i][j] = __builtin_amdgcn_wmma_f32_16x16x32_bf16(
            false, fa[i].v, false, fb[j].v, (short)0, acc[i][j], false, false);
  }

  // epilogue: D layout — lane holds col n, VGPR r holds row (lh*8 + r)
#pragma unroll
  for (int i = 0; i < 2; ++i) {
#pragma unroll
    for (int j = 0; j < 2; ++j) {
      int n = n0 + wn + j * 16 + l16;
#pragma unroll
      for (int r = 0; r < 8; ++r) {
        int m = m0 + wm + i * 16 + lh * 8 + r;
        if (m < M) {
          float v = acc[i][j][r];
          if (F32OUT) {
            if (n < Nreal) {
              v = fmaxf(v + bias[n], 0.0f);
              Cf[(size_t)m * Nreal + n] = v;
            }
          } else {
            Cb[(size_t)m * Npad + n] = (__bf16)fmaxf(v, 0.0f);
          }
        }
      }
    }
  }
}

// ---------------------------------------------------------------------------
extern "C" void kernel_launch(void* const* d_in, const int* in_sizes, int n_in,
                              void* d_out, int out_size, void* d_ws, size_t ws_size,
                              hipStream_t stream) {
  const float* f_atoms = (const float*)d_in[0];
  const float* f_bonds = (const float*)d_in[1];
  const int*   a2b     = (const int*)d_in[2];
  const int*   b2a     = (const int*)d_in[3];
  const int*   b2revb  = (const int*)d_in[4];
  const int*   a_scope = (const int*)d_in[5];
  const float* W_i     = (const float*)d_in[6];
  const float* W_h     = (const float*)d_in[7];
  const float* W_o     = (const float*)d_in[8];
  const float* b_o     = (const float*)d_in[9];
  float* out = (float*)d_out;

  char* ws = (char*)d_ws;
  size_t off = 0;
  auto take = [&](size_t bytes) -> char* {
    char* p = ws + off;
    off = (off + bytes + 255) & ~(size_t)255;
    return p;
  };

  __bf16* Wi_bf = (__bf16*)take((size_t)HP * K1P * 2);
  __bf16* Wh_bf = (__bf16*)take((size_t)HP * HP * 2);
  __bf16* Wo_bf = (__bf16*)take((size_t)HP * K3P * 2);
  __bf16* X0    = (__bf16*)take((size_t)N_BONDS * K1P * 2);
  __bf16* Mbuf0 = (__bf16*)take((size_t)N_BONDS * HP * 2);
  __bf16* Mbuf1 = (__bf16*)take((size_t)N_BONDS * HP * 2);
  __bf16* Xbuf  = (__bf16*)take((size_t)N_BONDS * HP * 2);  // also hosts X2
  float*  HID   = (float*)Mbuf1;  // fp32 hidden reuses dead msg1 buffer

  const int T = 256;
  auto blk = [](long long total, int t) { return (int)((total + t - 1) / t); };

  // weight conversion (tiny)
  pad_weights_bf16<<<blk((long long)HP * K1P, T), T, 0, stream>>>(
      W_i, HIDDEN, ATOM_FDIM + BOND_FDIM, Wi_bf, HP, K1P);
  pad_weights_bf16<<<blk((long long)HP * HP, T), T, 0, stream>>>(
      W_h, HIDDEN, HIDDEN, Wh_bf, HP, HP);
  pad_weights_bf16<<<blk((long long)HP * K3P, T), T, 0, stream>>>(
      W_o, HIDDEN, ATOM_FDIM + HIDDEN, Wo_bf, HP, K3P);

  // X0 = concat(f_atoms[b2a], f_bonds) -> bf16 padded
  build_x0<<<blk((long long)N_BONDS * K1P, T), T, 0, stream>>>(f_atoms, f_bonds, b2a, X0);

  dim3 gB((N_BONDS + 127) / 128, HP / 64);
  dim3 gA((N_ATOMS + 127) / 128, HP / 64);

  // message0 = relu(X0 @ Wi^T)
  gemm_wmma<false><<<gB, T, 0, stream>>>(X0, K1P, Wi_bf, N_BONDS, Mbuf0, HP,
                                         nullptr, nullptr, 0);

  // depth-1 message passes
  build_x1<<<blk((long long)N_BONDS * (HP / 8), T), T, 0, stream>>>(
      Mbuf0, a2b, b2a, b2revb, Xbuf);
  gemm_wmma<false><<<gB, T, 0, stream>>>(Xbuf, HP, Wh_bf, N_BONDS, Mbuf1, HP,
                                         nullptr, nullptr, 0);
  build_x1<<<blk((long long)N_BONDS * (HP / 8), T), T, 0, stream>>>(
      Mbuf1, a2b, b2a, b2revb, Xbuf);
  gemm_wmma<false><<<gB, T, 0, stream>>>(Xbuf, HP, Wh_bf, N_BONDS, Mbuf0, HP,
                                         nullptr, nullptr, 0);

  // X2 = concat(f_atoms, neighbor-sum(message2))
  build_x2<<<blk((long long)N_ATOMS * K3P, T), T, 0, stream>>>(f_atoms, Mbuf0, a2b, Xbuf);

  // atom_hiddens = relu(X2 @ Wo^T + b_o)  (fp32 out)
  gemm_wmma<true><<<gA, T, 0, stream>>>(Xbuf, K3P, Wo_bf, N_ATOMS, nullptr, HP,
                                        HID, b_o, HIDDEN);

  // per-molecule mean
  mol_mean<<<N_MOLS, 128, 0, stream>>>(HID, a_scope, out);
}
